// Model_64338610094318
// MI455X (gfx1250) — matrix operand order repaired and checked by reading, not compiled, not run on hardware
//
#include <hip/hip_runtime.h>
#include <hip/hip_bf16.h>

// ---------------------------------------------------------------------------
// Bidirectional LayerNorm-GRU encoder for MI455X (gfx1250, wave32, WMMA).
// GEMMs: bf16 inputs, fp32 accumulate via v_wmma_f32_16x16x32_bf16, with
// double-buffered GLOBAL_LOAD_ASYNC_TO_LDS staging (ASYNCcnt-tracked).
// LayerNorm / gate math in fp32.
// ---------------------------------------------------------------------------

typedef __bf16 bf16_t;
typedef __bf16 v16bf __attribute__((ext_vector_type(16)));
typedef float  v8f   __attribute__((ext_vector_type(8)));
typedef int    v4i   __attribute__((ext_vector_type(4)));
typedef __attribute__((address_space(3))) v4i* lds_v4i_t;

#define S_LEN 1024
#define B_DIM 64
#define E_DIM 512
#define H_DIM 512
#define G3    (3 * H_DIM)          // 1536
#define SB    (S_LEN * B_DIM)      // 65536

#define BM 64
#define BN 256
#define BK 32

#if __has_builtin(__builtin_amdgcn_global_load_async_to_lds_b128)
#define HAS_ASYNC_LDS 1
#else
#define HAS_ASYNC_LDS 0
#endif

#if __has_builtin(__builtin_amdgcn_s_wait_asynccnt)
#define WAIT_ASYNC(n) __builtin_amdgcn_s_wait_asynccnt(n)
#else
#define WAIT_ASYNC(n) asm volatile("s_wait_asynccnt " #n ::: "memory")
#endif

// ---------------- small utility kernels ----------------

__global__ void k_cvt_bf16(const float* __restrict__ src, bf16_t* __restrict__ dst, int n) {
    int i = blockIdx.x * blockDim.x + threadIdx.x;
    if (i < n) dst[i] = (bf16_t)src[i];
}

__global__ void k_embed(const int* __restrict__ src, const float* __restrict__ emb,
                        bf16_t* __restrict__ x) {
    long i   = (long)blockIdx.x * blockDim.x + threadIdx.x;   // over SB*E
    long row = i >> 9;                                        // / E_DIM
    int  e   = (int)(i & (E_DIM - 1));
    int  tok = src[row];
    float v  = (tok != 0) ? emb[(long)tok * E_DIM + e] : 0.0f;
    x[i] = (bf16_t)v;
}

__global__ void k_zero_h(float* __restrict__ hf, bf16_t* __restrict__ hb) {
    int i = blockIdx.x * blockDim.x + threadIdx.x;            // 2*B*H elems
    hf[i] = 0.0f;
    hb[i] = (bf16_t)0.0f;
}

// ---------------- WMMA fragment loaders (wave32) ----------------
// A-matrix 16x32 bf16: lane L(0..15)->M=L, K=0..15;
// lanes 16..31 -> same M, K=16..31; 2 K-values per VGPR (matches frag_ld_b).
__device__ __forceinline__ v16bf frag_ld_a(const bf16_t* base, int stride, int lane) {
    int r  = lane & 15;
    int kh = (lane >> 4) << 4;
    const bf16_t* p = base + r * stride + kh;
    v16bf f;
#pragma unroll
    for (int v = 0; v < 8; ++v) {
        int k = v << 1;
        f[2 * v]     = p[k];
        f[2 * v + 1] = p[k + 1];
    }
    return f;
}

// B-matrix 32x16 bf16: lane L(0..15)->N=L holds K=0..15 (2/VGPR);
// lanes 16..31 -> same N, K=16..31. Row here = weight row (N), contiguous K.
__device__ __forceinline__ v16bf frag_ld_b(const bf16_t* base, int stride, int lane) {
    int n  = lane & 15;
    int ks = (lane >> 4) << 4;
    const bf16_t* p = base + n * stride + ks;
    v16bf f;
#pragma unroll
    for (int v = 0; v < 8; ++v) {
        f[2 * v]     = p[2 * v];
        f[2 * v + 1] = p[2 * v + 1];
    }
    return f;
}

// ---------------- global -> LDS tile staging (5 x 16B per thread) ----------
// ga: this thread's 16B slice of the 64x32 A tile
// gb: this thread's W row slice (32 bf16 = 4 x 16B) of the 256x32 B tile
__device__ __forceinline__ void stage_tile(const bf16_t* __restrict__ ga,
                                           const bf16_t* __restrict__ gb,
                                           bf16_t* sA, bf16_t* sB,
                                           int aIdx, int tid)
{
#if HAS_ASYNC_LDS
    __builtin_amdgcn_global_load_async_to_lds_b128(
        (v4i*)ga, (lds_v4i_t)(sA + aIdx), 0, 0);
#pragma unroll
    for (int j = 0; j < 4; ++j)
        __builtin_amdgcn_global_load_async_to_lds_b128(
            (v4i*)(gb + j * 8), (lds_v4i_t)(sB + tid * BK + j * 8), 0, 0);
#else
    // fallback: batch all global reads, then all LDS writes
    uint4 a  = *(const uint4*)ga;
    uint4 b0 = *(const uint4*)(gb);
    uint4 b1 = *(const uint4*)(gb + 8);
    uint4 b2 = *(const uint4*)(gb + 16);
    uint4 b3 = *(const uint4*)(gb + 24);
    *(uint4*)(sA + aIdx) = a;
    uint4* sp = (uint4*)(sB + tid * BK);
    sp[0] = b0; sp[1] = b1; sp[2] = b2; sp[3] = b3;
#endif
}

// ---------------- generic batched GEMM: C[z] = A[z] (MxK) * W[z] (NxK)^T ----
// Block tile 64x256, 8 waves, each wave 32x64 (2x4 WMMA tiles).
// Double-buffered LDS with async-to-LDS copies overlapping WMMA compute.
__global__ void __launch_bounds__(256)
k_gemm_bf16nt(const bf16_t* __restrict__ Ab, const bf16_t* __restrict__ Wb,
              float* __restrict__ Cb, int M, int N, int K,
              long strideA, long strideW, long strideC,
              const float* __restrict__ bias, int fuse_tanh)
{
    const bf16_t* A = Ab + (long)blockIdx.z * strideA;
    const bf16_t* W = Wb + (long)blockIdx.z * strideW;
    float*        C = Cb + (long)blockIdx.z * strideC;

    __shared__ __align__(16) bf16_t sA[2][BM * BK];   //  2 x  4 KB
    __shared__ __align__(16) bf16_t sB[2][BN * BK];   //  2 x 16 KB

    const int tid  = threadIdx.x;
    const int lane = tid & 31;
    const int wave = tid >> 5;     // 0..7
    const int wm   = wave >> 2;    // 0..1  (32 rows each)
    const int wn   = wave & 3;     // 0..3  (64 cols each)

    const int m0 = blockIdx.y * BM;
    const int n0 = blockIdx.x * BN;

    // per-thread staging source addresses
    const int aIdx = tid * 8;                  // element offset into sA tile
    const bf16_t* gA = A + (long)(m0 + (aIdx >> 5)) * K + (aIdx & 31);
    const bf16_t* gB = W + (long)(n0 + tid) * K;

    v8f acc[2][4];
#pragma unroll
    for (int i = 0; i < 2; ++i)
#pragma unroll
        for (int j = 0; j < 4; ++j)
#pragma unroll
            for (int e = 0; e < 8; ++e) acc[i][j][e] = 0.0f;

    const int kTiles = K / BK;

    // prologue: stage tile 0 into buffer 0
    stage_tile(gA, gB, sA[0], sB[0], aIdx, tid);

    for (int kb = 0; kb < kTiles; ++kb) {
        const int cur = kb & 1;

        // stage next tile into the other buffer while we compute this one
        if (kb + 1 < kTiles)
            stage_tile(gA + (kb + 1) * BK, gB + (kb + 1) * BK,
                       sA[cur ^ 1], sB[cur ^ 1], aIdx, tid);

#if HAS_ASYNC_LDS
        // allow the 5 just-issued copies to remain in flight; older ones done
        if (kb + 1 < kTiles) WAIT_ASYNC(5); else WAIT_ASYNC(0);
#endif
        __syncthreads();   // publish buffer `cur` to all waves

        v16bf af[2];
#pragma unroll
        for (int mi = 0; mi < 2; ++mi)
            af[mi] = frag_ld_a(&sA[cur][(wm * 32 + mi * 16) * BK], BK, lane);
#pragma unroll
        for (int ni = 0; ni < 4; ++ni) {
            v16bf bfrag = frag_ld_b(&sB[cur][(wn * 64 + ni * 16) * BK], BK, lane);
#pragma unroll
            for (int mi = 0; mi < 2; ++mi)
                acc[mi][ni] = __builtin_amdgcn_wmma_f32_16x16x32_bf16(
                    false, af[mi], false, bfrag, (short)0, acc[mi][ni],
                    false, false);
        }
        __syncthreads();   // WAR: done reading `cur` before it is re-staged
    }

    // epilogue: C/D layout -> N = lane&15, M = r + 8*(lane>>4)
    const int cn = lane & 15;
    const int mh = lane >> 4;
#pragma unroll
    for (int mi = 0; mi < 2; ++mi) {
#pragma unroll
        for (int ni = 0; ni < 4; ++ni) {
            int nb = n0 + wn * 64 + ni * 16 + cn;
            float bv = bias ? bias[nb] : 0.0f;
#pragma unroll
            for (int r = 0; r < 8; ++r) {
                int mm = m0 + wm * 32 + mi * 16 + r + 8 * mh;
                float v = acc[mi][ni][r] + bv;
                if (fuse_tanh) v = tanhf(v);
                C[(long)mm * N + nb] = v;
            }
        }
    }
}

// ---------------- per-gate LayerNorm on xp rows (in place) ------------------
// One block per (row, gate) chunk of H=512; out = xn*g + be + b_ih.
__global__ void __launch_bounds__(256)
k_ln3(float* __restrict__ xp, const float* __restrict__ g,
      const float* __restrict__ be, const float* __restrict__ bih)
{
    long blk  = blockIdx.x;         // rows*3
    long row  = blk / 3;
    int  gate = (int)(blk % 3);
    float* p  = xp + row * G3 + (long)gate * H_DIM;
    int tid   = threadIdx.x;

    float x0 = p[tid], x1 = p[tid + 256];

    __shared__ float s1[256], s2[256];
    s1[tid] = x0 + x1;
    s2[tid] = x0 * x0 + x1 * x1;
    __syncthreads();
    for (int off = 128; off > 0; off >>= 1) {
        if (tid < off) { s1[tid] += s1[tid + off]; s2[tid] += s2[tid + off]; }
        __syncthreads();
    }
    float mu   = s1[0] * (1.0f / H_DIM);
    float var  = s2[0] * (1.0f / H_DIM) - mu * mu;
    float rinv = rsqrtf(var + 1e-5f);

    int j0 = gate * H_DIM + tid;
    p[tid]       = (x0 - mu) * rinv * g[j0]       + be[j0]       + bih[j0];
    p[tid + 256] = (x1 - mu) * rinv * g[j0 + 256] + be[j0 + 256] + bih[j0 + 256];
}

// ---------------- one recurrence step: LN(hp) + gates -----------------------
// grid = 2*B blocks (dir-major), 256 threads; each thread 2 hidden elems.
__global__ void __launch_bounds__(256)
k_gru_step(const float* __restrict__ hp_raw,   // [2][B][3H]
           const float* __restrict__ xp,       // [2][SB][3H] (post-LN+bias)
           float* __restrict__ hf,             // [2][B][H] fp32 state
           bf16_t* __restrict__ hb,            // [2][B][H] bf16 mirror
           bf16_t* __restrict__ y,             // [SB][2H] bf16 outputs
           float* __restrict__ hidden_out,     // [2][B][H] final states
           const float* __restrict__ ghh_f, const float* __restrict__ behh_f,
           const float* __restrict__ bhh_f,
           const float* __restrict__ ghh_b, const float* __restrict__ behh_b,
           const float* __restrict__ bhh_b,
           int step)
{
    int blk = blockIdx.x;
    int dir = blk >> 6;
    int b   = blk & 63;
    int tid = threadIdx.x;
    int t   = dir ? (S_LEN - 1 - step) : step;

    const float* hp   = hp_raw + ((long)dir * B_DIM + b) * G3;
    const float* ghh  = dir ? ghh_b  : ghh_f;
    const float* behh = dir ? behh_b : behh_f;
    const float* bhh  = dir ? bhh_b  : bhh_f;

    __shared__ float s1[256], s2[256];
    __shared__ float mu[3], rinv[3];

    for (int g = 0; g < 3; ++g) {
        float x0 = hp[g * H_DIM + tid];
        float x1 = hp[g * H_DIM + tid + 256];
        s1[tid] = x0 + x1;
        s2[tid] = x0 * x0 + x1 * x1;
        __syncthreads();
        for (int off = 128; off > 0; off >>= 1) {
            if (tid < off) { s1[tid] += s1[tid + off]; s2[tid] += s2[tid + off]; }
            __syncthreads();
        }
        if (tid == 0) {
            float m = s1[0] * (1.0f / H_DIM);
            float v = s2[0] * (1.0f / H_DIM) - m * m;
            mu[g]   = m;
            rinv[g] = rsqrtf(v + 1e-5f);
        }
        __syncthreads();
    }

    const float* xr   = xp + ((long)dir * SB + (long)t * B_DIM + b) * G3;
    long         hoff = ((long)dir * B_DIM + b) * H_DIM;
    long         yoff = ((long)t * B_DIM + b) * (2 * H_DIM) + (long)dir * H_DIM;

#pragma unroll
    for (int u = 0; u < 2; ++u) {
        int j = tid + u * 256;
        float hr = (hp[j]        - mu[0]) * rinv[0] * ghh[j]        + behh[j]        + bhh[j];
        float hz = (hp[512 + j]  - mu[1]) * rinv[1] * ghh[512 + j]  + behh[512 + j]  + bhh[512 + j];
        float hn = (hp[1024 + j] - mu[2]) * rinv[2] * ghh[1024 + j] + behh[1024 + j] + bhh[1024 + j];
        float r  = 1.0f / (1.0f + expf(-(xr[j]       + hr)));
        float z  = 1.0f / (1.0f + expf(-(xr[512 + j] + hz)));
        float n  = tanhf(xr[1024 + j] + r * hn);
        float hold = hf[hoff + j];
        float hnew = (1.0f - z) * n + z * hold;
        hf[hoff + j] = hnew;
        hb[hoff + j] = (bf16_t)hnew;
        y[yoff + j]  = (bf16_t)hnew;
        if (step == S_LEN - 1) hidden_out[hoff + j] = hnew;
    }
}

// ---------------------------------------------------------------------------

extern "C" void kernel_launch(void* const* d_in, const int* in_sizes, int n_in,
                              void* d_out, int out_size, void* d_ws, size_t ws_size,
                              hipStream_t stream)
{
    (void)in_sizes; (void)n_in; (void)out_size; (void)ws_size;

    const int*   src    = (const int*)  d_in[0];
    const float* emb    = (const float*)d_in[1];
    const float* Wih_f  = (const float*)d_in[2];
    const float* Whh_f  = (const float*)d_in[3];
    const float* bih_f  = (const float*)d_in[4];
    const float* bhh_f  = (const float*)d_in[5];
    const float* gih_f  = (const float*)d_in[6];
    const float* beih_f = (const float*)d_in[7];
    const float* ghh_f  = (const float*)d_in[8];
    const float* behh_f = (const float*)d_in[9];
    const float* Wih_b  = (const float*)d_in[10];
    const float* Whh_b  = (const float*)d_in[11];
    const float* bih_b  = (const float*)d_in[12];
    const float* bhh_b  = (const float*)d_in[13];
    const float* gih_b  = (const float*)d_in[14];
    const float* beih_b = (const float*)d_in[15];
    const float* ghh_b  = (const float*)d_in[16];
    const float* behh_b = (const float*)d_in[17];
    const float* Wout   = (const float*)d_in[18];
    const float* bout   = (const float*)d_in[19];

    char*  ws  = (char*)d_ws;
    size_t off = 0;
    auto take = [&](size_t bytes) -> char* {
        char* p = ws + off;
        off = (off + bytes + 255) & ~(size_t)255;
        return p;
    };

    bf16_t* xbf   = (bf16_t*)take((size_t)SB * E_DIM * 2);          //  64 MB
    bf16_t* WihB  = (bf16_t*)take((size_t)2 * G3 * E_DIM * 2);      //   3 MB
    bf16_t* WhhB  = (bf16_t*)take((size_t)2 * G3 * H_DIM * 2);      //   3 MB
    bf16_t* WoutB = (bf16_t*)take((size_t)H_DIM * 2 * H_DIM * 2);   //   1 MB
    float*  xp    = (float*) take((size_t)2 * SB * G3 * 4);         // 768 MB
    bf16_t* ybf   = (bf16_t*)take((size_t)SB * 2 * H_DIM * 2);      // 128 MB
    float*  hpraw = (float*) take((size_t)2 * B_DIM * G3 * 4);
    float*  hf32  = (float*) take((size_t)2 * B_DIM * H_DIM * 4);
    bf16_t* hbf   = (bf16_t*)take((size_t)2 * B_DIM * H_DIM * 2);

    // --- weight conversions to bf16 ---
    const int nW  = G3 * E_DIM;           // 786432 (E==H so Whh same size)
    const int nWo = H_DIM * 2 * H_DIM;    // 524288
    k_cvt_bf16<<<(nW  + 255) / 256, 256, 0, stream>>>(Wih_f, WihB,      nW);
    k_cvt_bf16<<<(nW  + 255) / 256, 256, 0, stream>>>(Wih_b, WihB + nW, nW);
    k_cvt_bf16<<<(nW  + 255) / 256, 256, 0, stream>>>(Whh_f, WhhB,      nW);
    k_cvt_bf16<<<(nW  + 255) / 256, 256, 0, stream>>>(Whh_b, WhhB + nW, nW);
    k_cvt_bf16<<<(nWo + 255) / 256, 256, 0, stream>>>(Wout,  WoutB,     nWo);

    // --- embedding gather (+ padding_idx mask) to bf16 ---
    k_embed<<<(SB * E_DIM) / 256, 256, 0, stream>>>(src, emb, xbf);

    // --- input projection, both directions in one batched launch ---
    k_gemm_bf16nt<<<dim3(G3 / BN, SB / BM, 2), 256, 0, stream>>>(
        xbf, WihB, xp, SB, G3, E_DIM,
        0L, (long)G3 * E_DIM, (long)SB * G3, nullptr, 0);

    // --- per-gate LayerNorm + LN affine + b_ih ---
    k_ln3<<<SB * 3, 256, 0, stream>>>(xp,                   gih_f, beih_f, bih_f);
    k_ln3<<<SB * 3, 256, 0, stream>>>(xp + (size_t)SB * G3, gih_b, beih_b, bih_b);

    // --- init hidden state ---
    k_zero_h<<<(2 * B_DIM * H_DIM) / 256, 256, 0, stream>>>(hf32, hbf);

    float* hidden_out = (float*)d_out + (size_t)SB * H_DIM;

    // --- sequential recurrence: fwd step t and bwd step S-1-t each iter ---
    for (int step = 0; step < S_LEN; ++step) {
        k_gemm_bf16nt<<<dim3(G3 / BN, 1, 2), 256, 0, stream>>>(
            hbf, WhhB, hpraw, B_DIM, G3, H_DIM,
            (long)B_DIM * H_DIM, (long)G3 * H_DIM, (long)B_DIM * G3,
            nullptr, 0);
        k_gru_step<<<2 * B_DIM, 256, 0, stream>>>(
            hpraw, xp, hf32, hbf, ybf, hidden_out,
            ghh_f, behh_f, bhh_f, ghh_b, behh_b, bhh_b, step);
    }

    // --- output projection: tanh(y @ W_out^T + b_out) -> d_out outputs ---
    k_gemm_bf16nt<<<dim3(H_DIM / BN, SB / BM, 1), 256, 0, stream>>>(
        ybf, WoutB, (float*)d_out, SB, H_DIM, 2 * H_DIM,
        0L, 0L, 0L, bout, 1);
}
